// CapsNet_64931315581476
// MI455X (gfx1250) — matrix operand order repaired and checked by reading, not compiled, not run on hardware
//
#include <hip/hip_runtime.h>
#include <hip/hip_bf16.h>

typedef __attribute__((ext_vector_type(16))) _Float16 v16h;
typedef __attribute__((ext_vector_type(8)))  _Float16 v8h;
typedef __attribute__((ext_vector_type(8)))  float    v8f;
typedef int caps_v4i __attribute__((vector_size(16)));

#define CAPS_EPS 1e-8f

// -------- CDNA5 async global->LDS staging (guarded; falls back to b128 copy) --------
#if defined(__AMDGCN__) && __has_builtin(__builtin_amdgcn_global_load_async_to_lds_b128)
#define CAPS_ASYNC_LDS 1
#else
#define CAPS_ASYNC_LDS 0
#endif

__device__ __forceinline__ void g2l_b128(_Float16* l, const _Float16* g) {
#if CAPS_ASYNC_LDS
  // LDS generic address low 32 bits == LDS offset (ISA 10.2); AS1 == flat VA.
  __builtin_amdgcn_global_load_async_to_lds_b128(
      (__attribute__((address_space(1))) caps_v4i*)(uintptr_t)g,
      (__attribute__((address_space(3))) caps_v4i*)(unsigned int)(uintptr_t)l,
      0, 0);
#else
  *(v8h*)l = *(const v8h*)g;
#endif
}

__device__ __forceinline__ void wait_staging() {
#if CAPS_ASYNC_LDS
#if __has_builtin(__builtin_amdgcn_s_wait_asynccnt)
  __builtin_amdgcn_s_wait_asynccnt(0);
#else
  asm volatile("s_wait_asynccnt 0" ::: "memory");
#endif
#endif
}

// ---------------- conversion / reorder kernels ----------------
__global__ void k_cvt_f16(const float* __restrict__ in, _Float16* __restrict__ out, int n) {
  int i = blockIdx.x * blockDim.x + threadIdx.x;
  if (i < n) out[i] = (_Float16)in[i];
}

// f32 [K,N] row-major -> f16 [N,K] row-major (B-transpose for GEMM staging)
__global__ void k_cvt_f16_T(const float* __restrict__ in, _Float16* __restrict__ out, int K, int N) {
  int i = blockIdx.x * blockDim.x + threadIdx.x;
  if (i < K * N) { int k = i / N, n = i - k * N; out[n * K + k] = (_Float16)in[i]; }
}

// conv1 weights [256,81] f32 -> f16 [256,96] zero-padded (16B-aligned rows)
__global__ void k_pad_conv1_w(const float* __restrict__ in, _Float16* __restrict__ out) {
  int i = blockIdx.x * blockDim.x + threadIdx.x;
  if (i >= 256 * 96) return;
  int n = i / 96, k = i - n * 96;
  out[i] = (k < 81) ? (_Float16)in[n * 81 + k] : (_Float16)0.f;
}

// primary-caps weights [OC=256][IC=256][9][9] f32 -> f16 [OC][(ky*9+kx)*256 + ic] (ic innermost)
__global__ void k_reorder_prim_w(const float* __restrict__ in, _Float16* __restrict__ out) {
  int i = blockIdx.x * blockDim.x + threadIdx.x;
  if (i >= 256 * 20736) return;
  int n = i / 20736, rem = i - n * 20736;
  int cell = rem >> 8, ic = rem & 255;
  int ky = cell / 9, kx = cell - ky * 9;
  out[i] = (_Float16)in[((n * 256 + ic) * 9 + ky) * 9 + kx];
}

// ---------------- WMMA fragment helpers (ISA 7.12.2 lane layouts) ----------------
__device__ __forceinline__ v16h load_a_frag(const _Float16* rowp, int kh) {
  v8h lo = *(const v8h*)(rowp + kh);
  v8h hi = *(const v8h*)(rowp + 16 + kh);
  v16h a;
  #pragma unroll
  for (int i = 0; i < 8; i++) { a[i] = lo[i]; a[8 + i] = hi[i]; }
  return a;
}
__device__ __forceinline__ v16h load_b_frag(const _Float16* colp, int kb) {
  v8h lo = *(const v8h*)(colp + kb);
  v8h hi = *(const v8h*)(colp + 16 + kb);
  v16h b;
  #pragma unroll
  for (int i = 0; i < 8; i++) { b[i] = lo[i]; b[8 + i] = hi[i]; }
  return b;
}

// ---------------- plain GEMM: C[M,N] = A[M,K]*B[K,N]+bias; Bt given as [N,K] f16 ----------------
// M mult of 64; K mult of 32. Used for decoder layers.
__launch_bounds__(128)
__global__ void k_gemm_wmma(const _Float16* __restrict__ A, const _Float16* __restrict__ Bt,
                            const float* __restrict__ bias,
                            _Float16* __restrict__ outH, float* __restrict__ outF,
                            int M, int N, int K, int relu)
{
  __shared__ __align__(16) _Float16 As[64][40];
  __shared__ __align__(16) _Float16 Bs[64][40];
  const int tid = threadIdx.x;
  const int lane = tid & 31, wv = tid >> 5;
  const int m0 = blockIdx.y * 64, n0 = blockIdx.x * 64;

  const int sr = tid & 63;          // staged row
  const int sh = (tid >> 6) * 16;   // halves [sh, sh+16)
  const _Float16* arow_g = A + (size_t)(m0 + sr) * K;
  const _Float16* brow_g = Bt + (size_t)(n0 + sr) * K;
  const bool bvalid = (n0 + sr) < N;

  v8f acc[4] = {};
  const int arow = (lane & 15) + wv * 16;
  const int akh  = (lane >> 4) * 8;
  const int bkb  = (lane >> 4) * 8;

  for (int k0 = 0; k0 < K; k0 += 32) {
    g2l_b128(&As[sr][sh],     arow_g + k0 + sh);
    g2l_b128(&As[sr][sh + 8], arow_g + k0 + sh + 8);
    if (bvalid) {
      g2l_b128(&Bs[sr][sh],     brow_g + k0 + sh);
      g2l_b128(&Bs[sr][sh + 8], brow_g + k0 + sh + 8);
    } else {
      v8h z = {};
      *(v8h*)&Bs[sr][sh] = z;
      *(v8h*)&Bs[sr][sh + 8] = z;
    }
    wait_staging();
    __syncthreads();
    v16h a = load_a_frag(&As[arow][0], akh);
    #pragma unroll
    for (int nb = 0; nb < 4; nb++) {
      v16h b = load_b_frag(&Bs[nb * 16 + (lane & 15)][0], bkb);
      acc[nb] = __builtin_amdgcn_wmma_f32_16x16x32_f16(false, a, false, b,
                                                       (short)0, acc[nb], false, false);
    }
    __syncthreads();
  }
  const int mrow = wv * 16 + 8 * (lane >> 4);
  const int ncol = lane & 15;
  for (int nb = 0; nb < 4; nb++) {
    int gn = n0 + nb * 16 + ncol;
    if (gn >= N) continue;
    float bv = bias[gn];
    #pragma unroll
    for (int r = 0; r < 8; r++) {
      int gm = m0 + mrow + r;
      float v = acc[nb][r] + bv;
      if (relu) v = v > 0.f ? v : 0.f;
      if (outH) outH[gm * N + gn] = (_Float16)v;
      if (outF) outF[gm * N + gn] = v;
    }
  }
}

// ---------------- conv1: generic implicit-im2col GEMM, NHWC f16 output [M][OC] ----------------
// Wt f16 [OC][Wstride] zero-padded; k order (ic,ky,kx). Only ~4% of total FLOPs.
__launch_bounds__(128)
__global__ void k_conv_gemm_wmma(const _Float16* __restrict__ in, const _Float16* __restrict__ Wt,
                                 const float* __restrict__ bias, _Float16* __restrict__ outH,
                                 int Bn, int IC, int IH, int IW,
                                 int OC, int OH, int OW, int stride, int Wstride, int relu)
{
  __shared__ __align__(16) _Float16 As[64][40];
  __shared__ __align__(16) _Float16 Bs[64][40];
  const int tid = threadIdx.x;
  const int lane = tid & 31, wv = tid >> 5;
  const int m0 = blockIdx.y * 64, n0 = blockIdx.x * 64;
  const int S = OH * OW;
  const int Ktot = IC * 81;

  const int sr = tid & 63;
  const int sh = (tid >> 6) * 16;
  const _Float16* brow_g = Wt + (size_t)(n0 + sr) * Wstride;

  v8f acc[4] = {};
  const int arow = (lane & 15) + wv * 16;
  const int akh  = (lane >> 4) * 8;
  const int bkb  = (lane >> 4) * 8;

  for (int k0 = 0; k0 < Wstride; k0 += 32) {
    // A: batched gather (issue all loads, then all LDS stores)
    _Float16 tmp[16];
    #pragma unroll
    for (int t2 = 0; t2 < 16; t2++) {
      int i = t2 * 128 + tid;
      int r = i >> 5, k = i & 31;
      int gk = k0 + k;
      _Float16 val = (_Float16)0.f;
      if (gk < Ktot) {
        int gm = m0 + r;
        int bidx = gm / S, sp = gm - bidx * S;
        int oy = sp / OW, ox = sp - oy * OW;
        int ic = gk / 81, r9 = gk - ic * 81;
        int ky = r9 / 9,  kx = r9 - ky * 9;
        val = in[((bidx * IC + ic) * IH + oy * stride + ky) * IW + ox * stride + kx];
      }
      tmp[t2] = val;
    }
    #pragma unroll
    for (int t2 = 0; t2 < 16; t2++) {
      int i = t2 * 128 + tid;
      As[i >> 5][i & 31] = tmp[t2];
    }
    // B: vector copy (rows are padded, 16B-aligned)
    g2l_b128(&Bs[sr][sh],     brow_g + k0 + sh);
    g2l_b128(&Bs[sr][sh + 8], brow_g + k0 + sh + 8);
    wait_staging();
    __syncthreads();
    v16h a = load_a_frag(&As[arow][0], akh);
    #pragma unroll
    for (int nb = 0; nb < 4; nb++) {
      v16h b = load_b_frag(&Bs[nb * 16 + (lane & 15)][0], bkb);
      acc[nb] = __builtin_amdgcn_wmma_f32_16x16x32_f16(false, a, false, b,
                                                       (short)0, acc[nb], false, false);
    }
    __syncthreads();
  }
  const int mrow = wv * 16 + 8 * (lane >> 4);
  const int ncol = lane & 15;
  for (int nb = 0; nb < 4; nb++) {
    int gn = n0 + nb * 16 + ncol;
    float bv = bias[gn];
    #pragma unroll
    for (int r = 0; r < 8; r++) {
      int gm = m0 + mrow + r;
      float v = acc[nb][r] + bv;
      if (relu) v = v > 0.f ? v : 0.f;
      outH[(size_t)gm * OC + gn] = (_Float16)v;   // NHWC
    }
  }
}

// ---------------- primary caps conv: NHWC input, double-buffered async staging ----------------
// in: f16 NHWC [Bn,IH,IW,C] (C mult of 64); Wt: f16 [OC][81*C], k=(ky*9+kx)*C+ic
// out: f32 NCHW [Bn,OC,OH,OW]. Tile 128(M) x 64(N) x 64(K); 256 threads = 8 waves.
__launch_bounds__(256)
__global__ void k_conv_nhwc_wmma(const _Float16* __restrict__ in, const _Float16* __restrict__ Wt,
                                 const float* __restrict__ bias, float* __restrict__ outF,
                                 int Bn, int IH, int IW, int C,
                                 int OC, int OH, int OW, int stride)
{
  __shared__ __align__(16) _Float16 As[2][128][72];
  __shared__ __align__(16) _Float16 Bs[2][64][72];
  const int tid = threadIdx.x;
  const int lane = tid & 31, wv = tid >> 5;
  const int m0 = blockIdx.y * 128, n0 = blockIdx.x * 64;
  const int S = OH * OW;
  const int K = 81 * C;
  const int nIter = K >> 6;

  // A staging: thread -> (row ar, halves [ah, ah+32))
  const int ar = tid & 127;
  const int ah = (tid >> 7) * 32;
  int gm = m0 + ar;
  int bI = gm / S, sp = gm - bI * S;
  int oy = sp / OW, ox = sp - oy * OW;
  const _Float16* inrow = in + ((size_t)(bI * IH + oy * stride) * IW + ox * stride) * C;
  // B staging: thread -> (row br, halves [bh, bh+16))
  const int br = tid & 63;
  const int bh = (tid >> 6) * 16;
  const _Float16* wrow = Wt + (size_t)(n0 + br) * K;

  v8f acc[4] = {};
  const int arow = (lane & 15) + wv * 16;
  const int akh  = (lane >> 4) * 8;
  const int bkb  = (lane >> 4) * 8;

  auto stage = [&](int buf, int k0) {
    int k = k0 + ah;
    int cell = k / C, ic = k - cell * C;     // 64-chunk stays inside one (ky,kx,C) run
    int ky = cell / 9, kx = cell - ky * 9;
    const _Float16* srcA = inrow + ((size_t)ky * IW + kx) * C + ic;
    #pragma unroll
    for (int j = 0; j < 4; j++)
      g2l_b128(&As[buf][ar][ah + j * 8], srcA + j * 8);
    const _Float16* srcB = wrow + k0 + bh;
    g2l_b128(&Bs[buf][br][bh],     srcB);
    g2l_b128(&Bs[buf][br][bh + 8], srcB + 8);
  };

  // prologue: fill buffer 0
  stage(0, 0);
  wait_staging();
  __syncthreads();

  for (int it = 0; it < nIter; ++it) {
    const int cur = it & 1;
    // issue async staging of the next tile; DMA overlaps the WMMAs below.
    // (end-of-previous-iteration barrier guarantees buf cur^1 was fully consumed)
    if (it + 1 < nIter) stage(cur ^ 1, (it + 1) << 6);
    #pragma unroll
    for (int ks = 0; ks < 64; ks += 32) {
      v16h a = load_a_frag(&As[cur][arow][ks], akh);
      #pragma unroll
      for (int nb = 0; nb < 4; nb++) {
        v16h b = load_b_frag(&Bs[cur][nb * 16 + (lane & 15)][ks], bkb);
        acc[nb] = __builtin_amdgcn_wmma_f32_16x16x32_f16(false, a, false, b,
                                                         (short)0, acc[nb], false, false);
      }
    }
    wait_staging();
    __syncthreads();           // next tile visible to all waves; buffers safe to swap
  }

  const int mrow = wv * 16 + 8 * (lane >> 4);
  const int ncol = lane & 15;
  for (int nb = 0; nb < 4; nb++) {
    int gn = n0 + nb * 16 + ncol;
    float bv = bias[gn];
    #pragma unroll
    for (int r = 0; r < 8; r++) {
      int g = m0 + mrow + r;
      int bb = g / S, ss = g - bb * S;
      outF[(size_t)(bb * OC + gn) * S + ss] = acc[nb][r] + bv;   // NCHW
    }
  }
}

// ---------------- squash primary capsules: p [B,1152,8] -> u ----------------
__global__ void k_squash(const float* __restrict__ p, float* __restrict__ u, int total) {
  int i = blockIdx.x * blockDim.x + threadIdx.x;
  if (i >= total) return;
  float x[8], nrm = 0.f;
  #pragma unroll
  for (int d = 0; d < 8; d++) { x[d] = p[i * 8 + d]; nrm += x[d] * x[d]; }
  float sc = nrm / (nrm + 1.f) / (sqrtf(nrm) + CAPS_EPS);
  #pragma unroll
  for (int d = 0; d < 8; d++) u[i * 8 + d] = x[d] * sc;
}

// ---------------- dynamic routing, one block per (b,c) ----------------
// b_logits[n,e] = u_hat[n,e] * V[e] where V accumulates v over iterations (b starts at 0).
__launch_bounds__(256)
__global__ void k_routing(const float* __restrict__ u, const float* __restrict__ Wr,
                          float* __restrict__ vout)
{
  const int b = blockIdx.x, c = blockIdx.y;
  __shared__ _Float16 uh[1152][16];
  __shared__ float red[256], red2[256];
  __shared__ float sV[16], sS[16], sMx[16];
  __shared__ float sScale;
  const int tid = threadIdx.x;

  for (int n = tid; n < 1152; n += 256) {
    float ud[8];
    #pragma unroll
    for (int d = 0; d < 8; d++) ud[d] = u[(b * 1152 + n) * 8 + d];
    const float* wp = Wr + ((size_t)(c * 1152 + n) * 8) * 16;
    #pragma unroll
    for (int e = 0; e < 16; e++) {
      float acc = 0.f;
      #pragma unroll
      for (int d = 0; d < 8; d++) acc += ud[d] * wp[d * 16 + e];
      uh[n][e] = (_Float16)acc;
    }
  }
  if (tid < 16) sV[tid] = 0.f;
  __syncthreads();

  const int e = tid & 15, ch = tid >> 4;
  for (int it = 0; it < 3; it++) {
    float Ve = sV[e];
    float mx = -3.4e38f;
    for (int n = ch; n < 1152; n += 16) mx = fmaxf(mx, (float)uh[n][e] * Ve);
    red[tid] = mx;
    __syncthreads();
    if (tid < 16) {
      float m = red[tid];
      for (int cc = 1; cc < 16; cc++) m = fmaxf(m, red[cc * 16 + tid]);
      sMx[tid] = m;
    }
    __syncthreads();
    float m = sMx[e], se = 0.f, su = 0.f;
    for (int n = ch; n < 1152; n += 16) {
      float uhv = (float)uh[n][e];
      float w = __expf(uhv * Ve - m);
      se += w; su += w * uhv;
    }
    red[tid] = se; red2[tid] = su;
    __syncthreads();
    if (tid < 16) {
      float den = 0.f, s = 0.f;
      for (int cc = 0; cc < 16; cc++) { den += red[cc * 16 + tid]; s += red2[cc * 16 + tid]; }
      sS[tid] = s / den;
    }
    __syncthreads();
    if (tid == 0) {
      float nrm = 0.f;
      for (int ee = 0; ee < 16; ee++) nrm += sS[ee] * sS[ee];
      sScale = nrm / (nrm + 1.f) / (sqrtf(nrm) + CAPS_EPS);
    }
    __syncthreads();
    if (tid < 16) {
      float v = sScale * sS[tid];
      if (it < 2) sV[tid] += v;
      else vout[(b * 10 + c) * 16 + tid] = v;
    }
    __syncthreads();
  }
}

// ---------------- class logits: softmax over C of ||v|| ----------------
__global__ void k_logits(const float* __restrict__ v, float* __restrict__ out) {
  int b = blockIdx.x, t = threadIdx.x;
  __shared__ float nrm[10];
  if (t < 10) {
    float s = 0.f;
    #pragma unroll
    for (int e = 0; e < 16; e++) { float x = v[(b * 10 + t) * 16 + e]; s += x * x; }
    nrm[t] = sqrtf(s);
  }
  __syncthreads();
  if (t < 10) {
    float m = nrm[0];
    for (int c = 1; c < 10; c++) m = fmaxf(m, nrm[c]);
    float den = 0.f;
    for (int c = 0; c < 10; c++) den += __expf(nrm[c] - m);
    out[b * 10 + t] = __expf(nrm[t] - m) / den;
  }
}

// ---------------- one-hot mask -> f16 decoder input [B,160] ----------------
__global__ void k_mask(const float* __restrict__ v, const int* __restrict__ y,
                       _Float16* __restrict__ dm, int total) {
  int i = blockIdx.x * blockDim.x + threadIdx.x;
  if (i >= total) return;
  int b = i / 160, r = i - b * 160, c = r >> 4;
  dm[i] = (c == y[b]) ? (_Float16)v[i] : (_Float16)0.f;
}

// ---------------- host side ----------------
static inline size_t alignup(size_t x) { return (x + 255) & ~(size_t)255; }

extern "C" void kernel_launch(void* const* d_in, const int* in_sizes, int n_in,
                              void* d_out, int out_size, void* d_ws, size_t ws_size,
                              hipStream_t stream) {
  const float* x   = (const float*)d_in[0];
  const int*   y   = (const int*)  d_in[1];
  const float* c1w = (const float*)d_in[2];
  const float* c1b = (const float*)d_in[3];
  const float* pw  = (const float*)d_in[4];
  const float* pb  = (const float*)d_in[5];
  const float* rW  = (const float*)d_in[6];
  const float* w1  = (const float*)d_in[7];
  const float* b1  = (const float*)d_in[8];
  const float* w2  = (const float*)d_in[9];
  const float* b2  = (const float*)d_in[10];
  const float* w3  = (const float*)d_in[11];
  const float* b3  = (const float*)d_in[12];

  float* out_logits = (float*)d_out;          // [256,10]
  float* out_recons = (float*)d_out + 2560;   // [256,784]

  uint8_t* ws = (uint8_t*)d_ws;
  size_t off = 0;
  auto carve = [&](size_t bytes) { uint8_t* p = ws + off; off += alignup(bytes); return p; };

  _Float16* xh   = (_Float16*)carve(256 * 784 * 2);               // x f16 (NCHW, IC=1)
  _Float16* c1wh = (_Float16*)carve(256 * 96 * 2);                // conv1 w f16 padded [256,96]
  _Float16* hh   = (_Float16*)carve((size_t)102400 * 256 * 2);    // conv1 out f16 NHWC [B*400,256]
  _Float16* pwt  = (_Float16*)carve((size_t)256 * 20736 * 2);     // prim w f16, k=(ky,kx,ic)
  float*    pout = (float*)   carve((size_t)256 * 9216 * 4);      // prim out f32 NCHW
  float*    u    = (float*)   carve((size_t)256 * 9216 * 4);      // squashed caps
  float*    vbuf = (float*)   carve(256 * 160 * 4);               // v [B,10,16]
  _Float16* dm   = (_Float16*)carve(256 * 160 * 2);               // masked decoder input
  _Float16* w1t  = (_Float16*)carve(512 * 160 * 2);
  _Float16* w2t  = (_Float16*)carve((size_t)1024 * 512 * 2);
  _Float16* w3t  = (_Float16*)carve((size_t)784 * 1024 * 2);
  _Float16* d1   = (_Float16*)carve(256 * 512 * 2);
  _Float16* d2   = (_Float16*)carve(256 * 1024 * 2);
  (void)ws_size; (void)n_in; (void)in_sizes; (void)out_size;

  // conversions / reorders
  k_cvt_f16<<<(200704 + 255) / 256, 256, 0, stream>>>(x, xh, 200704);
  k_pad_conv1_w<<<(256 * 96 + 255) / 256, 256, 0, stream>>>(c1w, c1wh);
  k_reorder_prim_w<<<(5308416 + 255) / 256, 256, 0, stream>>>(pw, pwt);
  k_cvt_f16_T<<<(160 * 512 + 255) / 256, 256, 0, stream>>>(w1, w1t, 160, 512);
  k_cvt_f16_T<<<(512 * 1024 + 255) / 256, 256, 0, stream>>>(w2, w2t, 512, 1024);
  k_cvt_f16_T<<<(1024 * 784 + 255) / 256, 256, 0, stream>>>(w3, w3t, 1024, 784);

  // conv1 + relu -> f16 NHWC  (M=102400, N=256, K=81 padded to 96)
  k_conv_gemm_wmma<<<dim3(256 / 64, 102400 / 64), 128, 0, stream>>>(
      xh, c1wh, c1b, hh, 256, 1, 28, 28, 256, 20, 20, 1, 96, 1);
  // primary caps conv -> f32 NCHW  (M=9216, N=256, K=20736) — dominant GEMM, async double-buffer
  k_conv_nhwc_wmma<<<dim3(256 / 64, 9216 / 128), 256, 0, stream>>>(
      hh, pwt, pb, pout, 256, 20, 20, 256, 256, 6, 6, 2);

  // squash
  k_squash<<<(256 * 1152 + 255) / 256, 256, 0, stream>>>(pout, u, 256 * 1152);
  // routing
  k_routing<<<dim3(256, 10), 256, 0, stream>>>(u, rW, vbuf);
  // class probabilities
  k_logits<<<256, 32, 0, stream>>>(vbuf, out_logits);
  // mask
  k_mask<<<(256 * 160 + 255) / 256, 256, 0, stream>>>(vbuf, y, dm, 256 * 160);

  // decoder GEMMs
  k_gemm_wmma<<<dim3(512 / 64, 256 / 64), 128, 0, stream>>>(dm, w1t, b1, d1, nullptr, 256, 512, 160, 1);
  k_gemm_wmma<<<dim3(1024 / 64, 256 / 64), 128, 0, stream>>>(d1, w2t, b2, d2, nullptr, 256, 1024, 512, 1);
  k_gemm_wmma<<<dim3((784 + 63) / 64, 256 / 64), 128, 0, stream>>>(d2, w3t, b3, nullptr, out_recons, 256, 784, 1024, 0);
}
